// SimpleMoE_79912161509676
// MI455X (gfx1250) — matrix operand order repaired and checked by reading, not compiled, not run on hardware
//
#include <hip/hip_runtime.h>
#include <hip/hip_bf16.h>

// ---------------------------------------------------------------------------
// SimpleMoE forward for MI455X (gfx1250, wave32, WMMA).
// GEMMs: v_wmma_f32_16x16x32_bf16, A staged via async global->LDS copies
// (ASYNCcnt, triple-buffered, one barrier per k-step), B streamed from
// global (bf16 weights are L2-resident: 65MB head / 33MB experts vs 192MB L2).
// ---------------------------------------------------------------------------

typedef __attribute__((ext_vector_type(16))) __bf16 bf16x16;
typedef __attribute__((ext_vector_type(8)))  __bf16 bf16x8;
typedef __attribute__((ext_vector_type(8)))  float  f32x8;
typedef __attribute__((ext_vector_type(4)))  int    i32x4;

// address-space-qualified pointers for the async-copy builtin
typedef __attribute__((address_space(1))) i32x4* gptr_i32x4;
typedef __attribute__((address_space(3))) i32x4* lptr_i32x4;

#define D_      1024
#define H2_     2048      // 2*D
#define E_      8
#define VOCAB_  32000
#define N_      4096      // B*T
#define EPS_    1e-5f

#define LDS_STRIDE 40     // 64x32 bf16 slab rows padded to 40 elems (80B)

#if defined(__has_builtin)
#if __has_builtin(__builtin_amdgcn_global_load_async_to_lds_b128)
#define HAVE_ASYNC_LDS 1
#endif
#if __has_builtin(__builtin_amdgcn_s_wait_asynccnt)
#define HAVE_WAIT_ASYNC 1
#endif
#endif

__device__ __forceinline__ void wait_async_le1() {
#if defined(HAVE_ASYNC_LDS)
#if defined(HAVE_WAIT_ASYNC)
    __builtin_amdgcn_s_wait_asynccnt(1);
#else
    asm volatile("s_wait_asynccnt 0x1" ::: "memory");
#endif
#endif
}
__device__ __forceinline__ void wait_async_le0() {
#if defined(HAVE_ASYNC_LDS)
#if defined(HAVE_WAIT_ASYNC)
    __builtin_amdgcn_s_wait_asynccnt(0);
#else
    asm volatile("s_wait_asynccnt 0x0" ::: "memory");
#endif
#endif
}

// Stage one 64x32 bf16 A-slab into LDS (one 16B chunk per thread, 256 thr).
__device__ __forceinline__ void stage_a_slab(
    const __bf16* __restrict__ A, int K, int m_block, int k0,
    int t, __bf16* dst /* generic ptr into __shared__ slab */)
{
    const int row = t >> 2;            // 0..63
    const int c16 = (t & 3) * 8;       // 0,8,16,24
    const __bf16* g = A + (size_t)(m_block + row) * K + k0 + c16;
    __bf16* l = dst + row * LDS_STRIDE + c16;
#if defined(HAVE_ASYNC_LDS)
    __builtin_amdgcn_global_load_async_to_lds_b128(
        (gptr_i32x4)(void*)g, (lptr_i32x4)(void*)l,
        0 /*imm offset*/, 0 /*cpol*/);
#else
    *(bf16x8*)l = *(const bf16x8*)g;
#endif
}

// ---------------- WMMA fragment loaders (layouts per ISA 7.12.2) -----------
// A: 16x32 bf16 tile read from the LDS slab. Lane l holds row (l&15),
// K-chunks [half*8, half*8+8) and [16+half*8, ...), half = l>>4.
__device__ __forceinline__ bf16x16
load_a_frag_lds(const __bf16* __restrict__ As, int m_local, int lane) {
    const int m    = m_local + (lane & 15);
    const int half = lane >> 4;
    const __bf16* p = As + m * LDS_STRIDE + half * 8;
    bf16x8 lo = *(const bf16x8*)(p);
    bf16x8 hi = *(const bf16x8*)(p + 16);
    return __builtin_shufflevector(lo, hi, 0,1,2,3,4,5,6,7,8,9,10,11,12,13,14,15);
}

// B: 32x16 bf16 tile, stored N-major as Bt[NN][K]. Lane l holds column (l&15),
// K-chunks [k0+half*8, +8) and [k0+16+half*8, +8), half = l>>4.
__device__ __forceinline__ bf16x16
load_b_frag(const __bf16* __restrict__ Bt, int ldb, int n0, int k0, int lane) {
    const int n    = n0 + (lane & 15);
    const int half = lane >> 4;
    const __bf16* p = Bt + (size_t)n * ldb + k0 + half * 8;
    bf16x8 lo = *(const bf16x8*)(p);
    bf16x8 hi = *(const bf16x8*)(p + 16);
    return __builtin_shufflevector(lo, hi, 0,1,2,3,4,5,6,7,8,9,10,11,12,13,14,15);
}

// One 32-deep k-step of the 32x64 wave tile: 2 A-frags (LDS), 4 B-frags
// (global), 8 WMMAs.
__device__ __forceinline__ void gemm_step(
    const __bf16* __restrict__ As_buf, int m_local, int lane,
    const __bf16* __restrict__ Bt, int K, int n_wave, int k0,
    f32x8 acc[2][4])
{
    bf16x16 a0 = load_a_frag_lds(As_buf, m_local,      lane);
    bf16x16 a1 = load_a_frag_lds(As_buf, m_local + 16, lane);
#pragma unroll
    for (int j = 0; j < 4; ++j) {
        bf16x16 b = load_b_frag(Bt, K, n_wave + j * 16, k0, lane);
        acc[0][j] = __builtin_amdgcn_wmma_f32_16x16x32_bf16(
            false, a0, false, b, (short)0, acc[0][j], false, false);
        acc[1][j] = __builtin_amdgcn_wmma_f32_16x16x32_bf16(
            false, a1, false, b, (short)0, acc[1][j], false, false);
    }
}

// ---------------- Generic WMMA GEMM: C[M,NN] = A[M,K] * Bt[NN,K]^T ----------
// MODE 0: head   -> outF[m*NN+n] = acc + bias[n]
// MODE 1: gemm1  -> outB[m*NN+n] = bf16(relu(acc + bias[n]))
// MODE 2: gemm2  -> atomicAdd(outF[m*NN+n], gates[m*E+expert]*(acc+bias[n]))
// Block: 256 threads = 8 waves (2 M x 4 N), block tile 64 x 256,
// wave tile 32 x 64 (2x4 accumulators of 16x16).
// A slab triple-buffered in LDS: one barrier per k-step; the slab written by
// iteration s was last read at s-2, separated by the s-1 barrier.
template <int MODE>
__global__ __launch_bounds__(256) void gemm_wmma_kernel(
    const __bf16* __restrict__ A, const __bf16* __restrict__ Bt,
    const float* __restrict__ bias,
    float* __restrict__ outF, __bf16* __restrict__ outB,
    const float* __restrict__ gates,
    int M, int NN, int K, int expert)
{
    __shared__ __bf16 As[3][64 * LDS_STRIDE];

    const int t      = threadIdx.x;
    const int lane   = t & 31;
    const int wave   = t >> 5;
    const int m_block = blockIdx.y * 64;
    const int m_local = (wave & 1) * 32;
    const int n_wave  = blockIdx.x * 256 + (wave >> 1) * 64;

    const f32x8 zero = {0.f, 0.f, 0.f, 0.f, 0.f, 0.f, 0.f, 0.f};
    f32x8 acc[2][4];
#pragma unroll
    for (int i = 0; i < 2; ++i)
#pragma unroll
        for (int j = 0; j < 4; ++j) acc[i][j] = zero;

    // prime buffer 0
    stage_a_slab(A, K, m_block, 0, t, As[0]);

    // steady state: branch-free body (stage next, wait current, barrier, MMA)
    int k0 = 0;
    for (; k0 + 32 < K; k0 += 32) {
        const int s = k0 >> 5;
        stage_a_slab(A, K, m_block, k0 + 32, t, As[(s + 1) % 3]);
        wait_async_le1();              // current slab's copy complete
        __syncthreads();               // slab visible; also fences s-2 reads
        __builtin_prefetch(Bt + (size_t)(n_wave + (lane & 15)) * K + k0 + 32, 0, 3);
        gemm_step(As[s % 3], m_local, lane, Bt, K, n_wave, k0, acc);
    }
    // peeled last k-step
    wait_async_le0();
    __syncthreads();
    gemm_step(As[(k0 >> 5) % 3], m_local, lane, Bt, K, n_wave, k0, acc);

    // Epilogue. C layout: VGPR r, lanes0-15 -> M = r, lanes16-31 -> M = 8+r.
    const int half = lane >> 4;
    const int nl   = lane & 15;

    float gv[2][8];
    if (MODE == 2) {
#pragma unroll
        for (int i = 0; i < 2; ++i)
#pragma unroll
            for (int r = 0; r < 8; ++r) {
                const int m = m_block + m_local + i * 16 + half * 8 + r;
                gv[i][r] = gates[(size_t)m * E_ + expert];
            }
    }

#pragma unroll
    for (int i = 0; i < 2; ++i) {
#pragma unroll
        for (int j = 0; j < 4; ++j) {
            const int n  = n_wave + j * 16 + nl;
            const float bv = bias[n];
#pragma unroll
            for (int r = 0; r < 8; ++r) {
                const int m = m_block + m_local + i * 16 + half * 8 + r;
                float v = acc[i][j][r] + bv;
                if (MODE == 0) {
                    outF[(size_t)m * NN + n] = v;
                } else if (MODE == 1) {
                    outB[(size_t)m * NN + n] = (__bf16)fmaxf(v, 0.f);
                } else {
                    atomicAdd(&outF[(size_t)m * NN + n], gv[i][r] * v);
                }
            }
        }
    }
}

// ---------------- Weight conversion / transpose kernels ---------------------
__global__ __launch_bounds__(256) void cvt_bf16_kernel(
    const float* __restrict__ in, __bf16* __restrict__ out, size_t n)
{
    size_t i = (size_t)blockIdx.x * blockDim.x + threadIdx.x;
    size_t stride = (size_t)gridDim.x * blockDim.x;
    for (; i < n; i += stride) out[i] = (__bf16)in[i];
}

// W1 [E, D, 2D] f32 -> W1t [E, 2D, D] bf16 (N-major for B fragments)
__global__ __launch_bounds__(256) void cvt_w1t_kernel(
    const float* __restrict__ W1, __bf16* __restrict__ W1t)
{
    size_t total = (size_t)E_ * H2_ * D_;
    size_t i = (size_t)blockIdx.x * blockDim.x + threadIdx.x;
    size_t stride = (size_t)gridDim.x * blockDim.x;
    for (; i < total; i += stride) {
        size_t e = i / ((size_t)H2_ * D_);
        size_t r = i % ((size_t)H2_ * D_);
        size_t j = r / D_;       // 0..2047 out row
        size_t k = r % D_;       // 0..1023
        W1t[i] = (__bf16)W1[e * (size_t)D_ * H2_ + k * H2_ + j];
    }
}

// W2 [E, 2D, D] f32 -> W2t [E, D, 2D] bf16
__global__ __launch_bounds__(256) void cvt_w2t_kernel(
    const float* __restrict__ W2, __bf16* __restrict__ W2t)
{
    size_t total = (size_t)E_ * D_ * H2_;
    size_t i = (size_t)blockIdx.x * blockDim.x + threadIdx.x;
    size_t stride = (size_t)gridDim.x * blockDim.x;
    for (; i < total; i += stride) {
        size_t e = i / ((size_t)D_ * H2_);
        size_t r = i % ((size_t)D_ * H2_);
        size_t j = r / H2_;      // 0..1023 out row
        size_t k = r % H2_;      // 0..2047
        W2t[i] = (__bf16)W2[e * (size_t)H2_ * D_ + k * D_ + j];
    }
}

// ---------------- Embed gather + router (softmax + top-2) -------------------
__global__ __launch_bounds__(256) void embed_gate_kernel(
    const int* __restrict__ x, const float* __restrict__ embedW,
    const float* __restrict__ gateW, const float* __restrict__ gateB,
    __bf16* __restrict__ h_bf16, float* __restrict__ gates,
    float* __restrict__ out_idx)
{
    const int n = blockIdx.x;
    const int t = threadIdx.x;
    const int tok = x[n];
    const float* row = embedW + (size_t)tok * D_;

    float acc[E_];
#pragma unroll
    for (int e = 0; e < E_; ++e) acc[e] = 0.f;

    for (int d = t; d < D_; d += 256) {
        float v = row[d];
        h_bf16[(size_t)n * D_ + d] = (__bf16)v;
#pragma unroll
        for (int e = 0; e < E_; ++e) acc[e] += v * gateW[e * D_ + d];
    }

    __shared__ float red[E_][256];
#pragma unroll
    for (int e = 0; e < E_; ++e) red[e][t] = acc[e];
    __syncthreads();
    for (int off = 128; off > 0; off >>= 1) {
        if (t < off) {
#pragma unroll
            for (int e = 0; e < E_; ++e) red[e][t] += red[e][t + off];
        }
        __syncthreads();
    }

    if (t == 0) {
        float lg[E_], p[E_];
        float mx = -3.4e38f;
#pragma unroll
        for (int e = 0; e < E_; ++e) {
            lg[e] = red[e][0] + gateB[e];
            mx = fmaxf(mx, lg[e]);
        }
        float s = 0.f;
#pragma unroll
        for (int e = 0; e < E_; ++e) { p[e] = __expf(lg[e] - mx); s += p[e]; }
        float inv = 1.f / s;
#pragma unroll
        for (int e = 0; e < E_; ++e) p[e] *= inv;

        int i1 = 0;
#pragma unroll
        for (int e = 1; e < E_; ++e) if (p[e] > p[i1]) i1 = e;
        int i2 = (i1 == 0) ? 1 : 0;
#pragma unroll
        for (int e = 0; e < E_; ++e)
            if (e != i1 && p[e] > p[i2]) i2 = e;

#pragma unroll
        for (int e = 0; e < E_; ++e)
            gates[(size_t)n * E_ + e] = (e == i1) ? p[i1] : (e == i2) ? p[i2] : 0.f;

        out_idx[(size_t)n * 2 + 0] = (float)i1;
        out_idx[(size_t)n * 2 + 1] = (float)i2;
    }
}

// ---------------- LayerNorm -> bf16 -----------------------------------------
__global__ __launch_bounds__(256) void layernorm_kernel(
    const float* __restrict__ y, const float* __restrict__ g,
    const float* __restrict__ b, __bf16* __restrict__ yn)
{
    const int n = blockIdx.x;
    const int t = threadIdx.x;
    const float* row = y + (size_t)n * D_;

    float v[4];
    float s = 0.f, s2 = 0.f;
#pragma unroll
    for (int i = 0; i < 4; ++i) {
        v[i] = row[t + i * 256];
        s  += v[i];
        s2 += v[i] * v[i];
    }

    __shared__ float rs[256], rs2[256];
    rs[t] = s; rs2[t] = s2;
    __syncthreads();
    for (int off = 128; off > 0; off >>= 1) {
        if (t < off) { rs[t] += rs[t + off]; rs2[t] += rs2[t + off]; }
        __syncthreads();
    }
    const float mu  = rs[0] * (1.f / D_);
    const float var = rs2[0] * (1.f / D_) - mu * mu;
    const float rstd = rsqrtf(var + EPS_);

#pragma unroll
    for (int i = 0; i < 4; ++i) {
        const int d = t + i * 256;
        yn[(size_t)n * D_ + d] = (__bf16)((v[i] - mu) * rstd * g[d] + b[d]);
    }
}

// ---------------- Launch orchestration --------------------------------------
extern "C" void kernel_launch(void* const* d_in, const int* in_sizes, int n_in,
                              void* d_out, int out_size, void* d_ws, size_t ws_size,
                              hipStream_t stream)
{
    (void)in_sizes; (void)n_in; (void)out_size; (void)ws_size;

    const int*   x       = (const int*)  d_in[0];
    const float* embed_W = (const float*)d_in[1];
    const float* gate_W  = (const float*)d_in[2];
    const float* gate_b  = (const float*)d_in[3];
    const float* W1      = (const float*)d_in[4];
    const float* b1      = (const float*)d_in[5];
    const float* W2      = (const float*)d_in[6];
    const float* b2      = (const float*)d_in[7];
    const float* ln_g    = (const float*)d_in[8];
    const float* ln_b    = (const float*)d_in[9];
    const float* head_W  = (const float*)d_in[10];
    const float* head_b  = (const float*)d_in[11];

    float* logits  = (float*)d_out;                              // [N, VOCAB]
    float* out_idx = (float*)d_out + (size_t)N_ * VOCAB_;        // [N, 2]

    // Workspace layout (bytes, 256-aligned)
    char* ws = (char*)d_ws;
    __bf16* W1t    = (__bf16*)(ws + 0);                          // 33,554,432
    __bf16* W2t    = (__bf16*)(ws + 33554432u);                  // 33,554,432
    __bf16* headWb = (__bf16*)(ws + 67108864u);                  // 65,536,000
    __bf16* h_bf16 = (__bf16*)(ws + 132644864u);                 //  8,388,608
    float*  gates  = (float*) (ws + 141033472u);                 //    131,072
    __bf16* a_ws   = (__bf16*)(ws + 141164544u);                 // 16,777,216
    float*  y      = (float*) (ws + 157941760u);                 // 16,777,216
    __bf16* yn     = (__bf16*)(ws + 174718976u);                 //  8,388,608

    // 0) zero the combine buffer (graph-capturable memset)
    (void)hipMemsetAsync(y, 0, (size_t)N_ * D_ * sizeof(float), stream);

    // 1) weight conversions (f32 -> bf16, N-major for B fragments)
    cvt_w1t_kernel<<<2048, 256, 0, stream>>>(W1, W1t);
    cvt_w2t_kernel<<<2048, 256, 0, stream>>>(W2, W2t);
    cvt_bf16_kernel<<<4096, 256, 0, stream>>>(head_W, headWb,
                                              (size_t)VOCAB_ * D_);

    // 2) embed + router softmax + top-2
    embed_gate_kernel<<<N_, 256, 0, stream>>>(x, embed_W, gate_W, gate_b,
                                              h_bf16, gates, out_idx);

    // 3) MoE: dense over experts, gate-weighted combine (atomic f32 add)
    for (int e = 0; e < E_; ++e) {
        gemm_wmma_kernel<1><<<dim3(H2_ / 256, N_ / 64), 256, 0, stream>>>(
            h_bf16, W1t + (size_t)e * H2_ * D_, b1 + (size_t)e * H2_,
            nullptr, a_ws, nullptr, N_, H2_, D_, e);
        gemm_wmma_kernel<2><<<dim3(D_ / 256, N_ / 64), 256, 0, stream>>>(
            a_ws, W2t + (size_t)e * D_ * H2_, b2 + (size_t)e * D_,
            y, nullptr, gates, N_, D_, H2_, e);
    }

    // 4) LayerNorm -> bf16 activations for the head
    layernorm_kernel<<<N_, 256, 0, stream>>>(y, ln_g, ln_b, yn);

    // 5) vocab head: [N,1024] x [32000,1024]^T -> f32 logits
    gemm_wmma_kernel<0><<<dim3(VOCAB_ / 256, N_ / 64), 256, 0, stream>>>(
        yn, headWb, head_b, logits, nullptr, nullptr, N_, VOCAB_, D_, 0);
}
